// TiledMemoryMOE_67242007986680
// MI455X (gfx1250) — compile-verified, matrix operand order checked by reading, not run on hardware
//
#include <hip/hip_runtime.h>
#include <hip/hip_bf16.h>

typedef __bf16 bf16;
typedef bf16  v8bf  __attribute__((ext_vector_type(8)));
typedef bf16  v16bf __attribute__((ext_vector_type(16)));
typedef float v8f   __attribute__((ext_vector_type(8)));
typedef unsigned int u32x4 __attribute__((ext_vector_type(4)));
typedef unsigned int u32x8 __attribute__((ext_vector_type(8)));

#define D_DIM  1024
#define T_TOK  4096
#define NEXP   16
#define FDIM   512
#define FSDIM  1024
#define LDB    40   // LDS row stride in bf16 elems: 80B = 64B data + 16B pad (16B aligned)

__device__ __forceinline__ v16bf cat8(v8bf a, v8bf b) {
  return __builtin_shufflevector(a, b, 0,1,2,3,4,5,6,7,8,9,10,11,12,13,14,15);
}
__device__ __forceinline__ v8f v8f_zero() {
  v8f z;
#pragma unroll
  for (int i = 0; i < 8; ++i) z[i] = 0.f;
  return z;
}

// ---- CDNA5 async global->LDS copy (ASYNCcnt path), 16B per lane ----------
__device__ __forceinline__ void async_copy_b128(void* lds_dst, const void* gsrc) {
  unsigned l = (unsigned)(unsigned long long)lds_dst;  // low 32b of flat = LDS addr
  asm volatile("global_load_async_to_lds_b128 %0, %1, off"
               :: "v"(l), "v"(gsrc) : "memory");
}
__device__ __forceinline__ void wait_async_0() {
  asm volatile("s_wait_asynccnt 0x0" ::: "memory");
}
__device__ __forceinline__ void wait_async_1() {
  asm volatile("s_wait_asynccnt 0x1" ::: "memory");
}

// ---- CDNA5 Tensor Data Mover: 2D tile (rows x 32 bf16) global->LDS -------
// D# group1 for a 2D bf16 tile: tile_dim0=32 elems (64B rows), tile_dim1=rows,
// global row stride = kd elems; LDS padding 16B after every 64B -> LDB stride.
__device__ __forceinline__ u32x8 tdm_g1_2d(unsigned kd, unsigned rows) {
  u32x8 g1;
  g1[0] = (1u << 16)        // data_size = 1 -> 2 bytes
        | (1u << 20)        // pad_enable
        | (3u << 22)        // pad_interval = 3 -> every 16 DWORDs (64B)
        | (3u << 25);       // pad_amount   = 3 -> 4 DWORDs (16B)
  g1[1] = (kd & 0xFFFFu) << 16;        // tensor_dim0[15:0]   (bits 48..63)
  g1[2] = (kd >> 16) | (rows << 16);   // tensor_dim0[31:16], tensor_dim1[15:0]
  g1[3] = (32u << 16);                 // tensor_dim1[31:16]=0, tile_dim0=32
  g1[4] = rows;                        // tile_dim1=rows, tile_dim2=0
  g1[5] = kd;                          // tensor_dim0_stride[31:0]
  g1[6] = 0;                           // stride hi, dim1_stride lo (unused, 2D)
  g1[7] = 0;
  return g1;
}
__device__ __forceinline__ void tdm_load_2d(unsigned lds_addr, const void* gaddr,
                                            u32x8 g1) {
  unsigned long long ga = (unsigned long long)gaddr;
  u32x4 g0;
  g0[0] = 1u;                                   // count=1, no gather/restore
  g0[1] = lds_addr;                             // lds_addr (bytes)
  g0[2] = (unsigned)ga;                         // global_addr[31:0]
  g0[3] = ((unsigned)(ga >> 32) & 0x01FFFFFFu)  // global_addr[56:32]
        | (2u << 30);                           // type = 2 ("image")
  asm volatile("tensor_load_to_lds %0, %1" :: "s"(g0), "s"(g1) : "memory");
}

// ---------------------------------------------------------------- utilities
__global__ void moe_zero_cnt(int* cnt) {
  if (threadIdx.x < NEXP) cnt[threadIdx.x] = 0;
}

__global__ __launch_bounds__(256)
void moe_cvt(const float* __restrict__ x, bf16* __restrict__ xb, int n) {
  int i = blockIdx.x * 256 + threadIdx.x;
  if (i < n) xb[i] = (bf16)x[i];
}

// transpose + fp32->bf16: src [R][C] row-major -> dst [C][R] bf16.
__global__ __launch_bounds__(256)
void moe_tcvt(const float* __restrict__ src, bf16* __restrict__ dst, int R, int C) {
  __shared__ float tile[32][33];
  const int tx = threadIdx.x, ty = threadIdx.y;
  const size_t bofs = (size_t)blockIdx.z * R * C;
  const int c0 = blockIdx.x * 32, r0 = blockIdx.y * 32;
#pragma unroll
  for (int i = 0; i < 4; ++i)
    tile[ty * 4 + i][tx] = src[bofs + (size_t)(r0 + ty * 4 + i) * C + c0 + tx];
  __syncthreads();
#pragma unroll
  for (int i = 0; i < 4; ++i)
    dst[bofs + (size_t)(c0 + ty * 4 + i) * R + r0 + tx] = (bf16)tile[tx][ty * 4 + i];
}

// ------------------------------------------------- router: softmax + top-2
__global__ __launch_bounds__(128)
void moe_router(const float* __restrict__ x, const float* __restrict__ Wr,
                int* __restrict__ cnt, int* __restrict__ tokIdx,
                float* __restrict__ tokW) {
  const int t = blockIdx.x;
  const int tid = threadIdx.x;
  float acc[NEXP];
#pragma unroll
  for (int e = 0; e < NEXP; ++e) acc[e] = 0.f;
  const float* xr = x + (size_t)t * D_DIM;
  for (int d = tid; d < D_DIM; d += 128) {
    float xv = xr[d];
    const float* w = Wr + (size_t)d * NEXP;
#pragma unroll
    for (int e = 0; e < NEXP; ++e) acc[e] += xv * w[e];
  }
  __shared__ float red[128 * NEXP];
#pragma unroll
  for (int e = 0; e < NEXP; ++e) red[tid * NEXP + e] = acc[e];
  __syncthreads();
  for (int s = 64; s > 0; s >>= 1) {
    if (tid < s) {
#pragma unroll
      for (int e = 0; e < NEXP; ++e)
        red[tid * NEXP + e] += red[(tid + s) * NEXP + e];
    }
    __syncthreads();
  }
  if (tid == 0) {
    float mx = -1e30f;
#pragma unroll
    for (int e = 0; e < NEXP; ++e) mx = fmaxf(mx, red[e]);
    float p[NEXP], sum = 0.f;
#pragma unroll
    for (int e = 0; e < NEXP; ++e) { p[e] = __expf(red[e] - mx); sum += p[e]; }
    float inv = 1.f / sum;
    for (int k = 0; k < 2; ++k) {
      int best = 0; float bv = -1.f;
#pragma unroll
      for (int e = 0; e < NEXP; ++e)
        if (p[e] > bv) { bv = p[e]; best = e; }
      int slot = atomicAdd(&cnt[best], 1);
      tokIdx[best * T_TOK + slot] = t;
      tokW[best * T_TOK + slot] = bv * inv;
      p[best] = -2.f;
    }
  }
}

__global__ void moe_scan(const int* __restrict__ cnt, int* __restrict__ offs) {
  if (threadIdx.x == 0) {
    int s = 0;
    for (int e = 0; e < NEXP; ++e) { offs[e] = s; s += cnt[e]; }
  }
}

// --------------------------------------------- gate+up GEMM fused w/ SiLU
// Weights pre-transposed bf16 [N][Kd]. B tiles double-buffered in LDS via
// the Tensor Data Mover (one 2D descriptor per matrix per K-step, wave 0).
__global__ __launch_bounds__(256)
void moe_gateup(const bf16* __restrict__ X, const bf16* __restrict__ WgTAll,
                const bf16* __restrict__ WuTAll, bf16* __restrict__ H,
                const int* __restrict__ tokIdx, const int* __restrict__ cnt,
                const int* __restrict__ offs, int N, int Kd, int lda) {
  const int e = blockIdx.z;
  const int Me = cnt ? cnt[e] : T_TOK;
  const int mBase = blockIdx.y * 128;
  if (mBase >= Me) return;
  const int nBase = blockIdx.x * 64;
  const int hBase = offs ? offs[e] : 0;
  const bf16* WgT = WgTAll + (size_t)e * N * Kd;
  const bf16* WuT = WuTAll + (size_t)e * N * Kd;

  const int tid = threadIdx.x;
  const int wave = tid >> 5, lane = tid & 31;
  const int lm = lane & 15;
  const int kLo = (lane < 16) ? 0 : 8;   // ISA 16-bit A/B fragment K split
  const int wm = mBase + wave * 16;
  const int r = wm + lm;
  const int rc = (r < Me) ? r : (Me - 1);
  const int tok = tokIdx ? tokIdx[e * T_TOK + rc] : rc;
  const bf16* aRow = X + (size_t)tok * lda + kLo;

  __shared__ bf16 lb[2][2][64 * LDB];   // [buf][mat][64 n-rows x LDB]

  v8f accG[4], accU[4];
#pragma unroll
  for (int s = 0; s < 4; ++s) { accG[s] = v8f_zero(); accU[s] = v8f_zero(); }

  const u32x8 g1 = tdm_g1_2d((unsigned)Kd, 64u);
  const bf16* gTile = WgT + (size_t)nBase * Kd;   // k advances along rows
  const bf16* uTile = WuT + (size_t)nBase * Kd;
  const unsigned ldsG[2] = { (unsigned)(unsigned long long)&lb[0][0][0],
                             (unsigned)(unsigned long long)&lb[1][0][0] };
  const unsigned ldsU[2] = { (unsigned)(unsigned long long)&lb[0][1][0],
                             (unsigned)(unsigned long long)&lb[1][1][0] };

  const int nIter = Kd / 32;
  if (wave == 0) {                      // prologue: tile 0 -> buf 0
    tdm_load_2d(ldsG[0], gTile, g1);
    tdm_load_2d(ldsU[0], uTile, g1);
  }

  for (int i = 0; i < nIter; ++i) {
    const int k0 = i * 32;
    if (wave == 0) {
      if (i + 1 < nIter) {              // issue next tile, then wait current
        const int b = (i + 1) & 1;
        tdm_load_2d(ldsG[b], gTile + k0 + 32, g1);
        tdm_load_2d(ldsU[b], uTile + k0 + 32, g1);
        __builtin_amdgcn_s_wait_tensorcnt(2);   // TDM completes in order
      } else {
        __builtin_amdgcn_s_wait_tensorcnt(0);
      }
    }
    __syncthreads();   // tile i visible to all waves

    v8bf a0 = *(const v8bf*)(aRow + k0);
    v8bf a1 = *(const v8bf*)(aRow + k0 + 16);
    v16bf A = cat8(a0, a1);
    const bf16* lbG = &lb[i & 1][0][0];
    const bf16* lbU = &lb[i & 1][1][0];
#pragma unroll
    for (int s = 0; s < 4; ++s) {
      const bf16* bg = &lbG[(s * 16 + lm) * LDB + kLo];
      const bf16* bu = &lbU[(s * 16 + lm) * LDB + kLo];
      v16bf Bg = cat8(*(const v8bf*)bg, *(const v8bf*)(bg + 16));
      v16bf Bu = cat8(*(const v8bf*)bu, *(const v8bf*)(bu + 16));
      accG[s] = __builtin_amdgcn_wmma_f32_16x16x32_bf16(
          false, A, false, Bg, (short)0, accG[s], false, false);
      accU[s] = __builtin_amdgcn_wmma_f32_16x16x32_bf16(
          false, A, false, Bu, (short)0, accU[s], false, false);
    }
    __syncthreads();   // all waves done reading buf i&1 before refill
  }

  // epilogue: SiLU(g)*u -> bf16 H. C layout: VGPR j -> m=j (lanes<16) / j+8.
#pragma unroll
  for (int s = 0; s < 4; ++s) {
#pragma unroll
    for (int j = 0; j < 8; ++j) {
      int m = (lane < 16) ? j : (j + 8);
      int rr = wm + m;
      if (rr < Me) {
        float g = accG[s][j];
        float h = (g / (1.f + __expf(-g))) * accU[s][j];
        H[(size_t)(hBase + rr) * N + nBase + s * 16 + lm] = (bf16)h;
      }
    }
  }
}

// --------------------------------------- down projection + scatter/combine
// B tiles via per-lane global_load_async_to_lds_b128 (ASYNCcnt pipeline).
__global__ __launch_bounds__(256)
void moe_down(const bf16* __restrict__ H, const bf16* __restrict__ WdTAll,
              float* __restrict__ out, const int* __restrict__ tokIdx,
              const int* __restrict__ cnt, const int* __restrict__ offs,
              const float* __restrict__ tokW, int N, int Kd) {
  const int e = blockIdx.z;
  const int Me = cnt ? cnt[e] : T_TOK;
  const int mBase = blockIdx.y * 128;
  if (mBase >= Me) return;
  const int nBase = blockIdx.x * 64;
  const int hBase = offs ? offs[e] : 0;
  const bf16* WdT = WdTAll + (size_t)e * N * Kd;

  const int tid = threadIdx.x;
  const int wave = tid >> 5, lane = tid & 31;
  const int lm = lane & 15;
  const int kLo = (lane < 16) ? 0 : 8;
  const int wm = mBase + wave * 16;
  const int r = wm + lm;
  const int rc = (r < Me) ? r : (Me - 1);
  const bf16* aRow = H + (size_t)(hBase + rc) * Kd + kLo;

  __shared__ bf16 lb[2][64 * LDB];

  v8f acc[4];
#pragma unroll
  for (int s = 0; s < 4; ++s) acc[s] = v8f_zero();

  const int nn = tid >> 2;
  const int kc = (tid & 3) * 8;
  const bf16* dsrc = WdT + (size_t)(nBase + nn) * Kd + kc;
  bf16* ldst = &lb[0][nn * LDB + kc];
  const size_t bufStride = (size_t)64 * LDB;

  const int nIter = Kd / 32;
  async_copy_b128(ldst, dsrc);

  for (int i = 0; i < nIter; ++i) {
    const int k0 = i * 32;
    if (i + 1 < nIter) {
      async_copy_b128(ldst + ((i + 1) & 1) * bufStride, dsrc + k0 + 32);
      wait_async_1();
    } else {
      wait_async_0();
    }
    __syncthreads();

    v8bf a0 = *(const v8bf*)(aRow + k0);
    v8bf a1 = *(const v8bf*)(aRow + k0 + 16);
    v16bf A = cat8(a0, a1);
    const bf16* lbC = &lb[i & 1][0];
#pragma unroll
    for (int s = 0; s < 4; ++s) {
      const bf16* bp = &lbC[(s * 16 + lm) * LDB + kLo];
      v16bf Bf = cat8(*(const v8bf*)bp, *(const v8bf*)(bp + 16));
      acc[s] = __builtin_amdgcn_wmma_f32_16x16x32_bf16(
          false, A, false, Bf, (short)0, acc[s], false, false);
    }
    __syncthreads();
  }

#pragma unroll
  for (int s = 0; s < 4; ++s) {
#pragma unroll
    for (int j = 0; j < 8; ++j) {
      int m = (lane < 16) ? j : (j + 8);
      int rr = wm + m;
      if (rr < Me) {
        float v = acc[s][j];
        int col = nBase + s * 16 + lm;
        if (tokIdx) {
          int t = tokIdx[e * T_TOK + rr];
          atomicAdd(&out[(size_t)t * N + col], v * tokW[e * T_TOK + rr]);
        } else {
          out[(size_t)rr * N + col] = v;
        }
      }
    }
  }
}

// ------------------------------------------------------------------ launch
extern "C" void kernel_launch(void* const* d_in, const int* in_sizes, int n_in,
                              void* d_out, int out_size, void* d_ws, size_t ws_size,
                              hipStream_t stream) {
  const float* x  = (const float*)d_in[0];
  const float* Wr = (const float*)d_in[1];
  const float* Wg = (const float*)d_in[2];
  const float* Wu = (const float*)d_in[3];
  const float* Wd = (const float*)d_in[4];
  const float* Sg = (const float*)d_in[5];
  const float* Su = (const float*)d_in[6];
  const float* Sd = (const float*)d_in[7];
  float* out = (float*)d_out;

  char* ws = (char*)d_ws;
  size_t off = 0;
  bf16* xb = (bf16*)(ws + off); off += (size_t)T_TOK * D_DIM * 2;        // 8 MB
  bf16* Hs = (bf16*)(ws + off); off += (size_t)T_TOK * FSDIM * 2;       // 8 MB
  bf16* Hr = (bf16*)(ws + off); off += (size_t)(2 * T_TOK) * FDIM * 2;  // 8 MB
  bf16* WgT = (bf16*)(ws + off); off += (size_t)NEXP * FDIM * D_DIM * 2;  // 16 MB
  bf16* WuT = (bf16*)(ws + off); off += (size_t)NEXP * FDIM * D_DIM * 2;  // 16 MB
  bf16* WdT = (bf16*)(ws + off); off += (size_t)NEXP * D_DIM * FDIM * 2;  // 16 MB
  bf16* SgT = (bf16*)(ws + off); off += (size_t)FSDIM * D_DIM * 2;        // 2 MB
  bf16* SuT = (bf16*)(ws + off); off += (size_t)FSDIM * D_DIM * 2;        // 2 MB
  bf16* SdT = (bf16*)(ws + off); off += (size_t)D_DIM * FSDIM * 2;        // 2 MB
  int*   cnt    = (int*)(ws + off); off += 64;
  int*   offs   = (int*)(ws + off); off += 64;
  int*   tokIdx = (int*)(ws + off); off += (size_t)NEXP * T_TOK * 4;
  float* tokW   = (float*)(ws + off); off += (size_t)NEXP * T_TOK * 4;

  hipLaunchKernelGGL(moe_zero_cnt, dim3(1), dim3(32), 0, stream, cnt);
  hipLaunchKernelGGL(moe_cvt, dim3((T_TOK * D_DIM) / 256), dim3(256), 0, stream,
                     x, xb, T_TOK * D_DIM);
  hipLaunchKernelGGL(moe_router, dim3(T_TOK), dim3(128), 0, stream,
                     x, Wr, cnt, tokIdx, tokW);
  hipLaunchKernelGGL(moe_scan, dim3(1), dim3(32), 0, stream, cnt, offs);

  // transpose+convert all weights to bf16 [N][K]
  hipLaunchKernelGGL(moe_tcvt, dim3(FDIM / 32, D_DIM / 32, NEXP), dim3(32, 8), 0, stream,
                     Wg, WgT, D_DIM, FDIM);
  hipLaunchKernelGGL(moe_tcvt, dim3(FDIM / 32, D_DIM / 32, NEXP), dim3(32, 8), 0, stream,
                     Wu, WuT, D_DIM, FDIM);
  hipLaunchKernelGGL(moe_tcvt, dim3(D_DIM / 32, FDIM / 32, NEXP), dim3(32, 8), 0, stream,
                     Wd, WdT, FDIM, D_DIM);
  hipLaunchKernelGGL(moe_tcvt, dim3(FSDIM / 32, D_DIM / 32, 1), dim3(32, 8), 0, stream,
                     Sg, SgT, D_DIM, FSDIM);
  hipLaunchKernelGGL(moe_tcvt, dim3(FSDIM / 32, D_DIM / 32, 1), dim3(32, 8), 0, stream,
                     Su, SuT, D_DIM, FSDIM);
  hipLaunchKernelGGL(moe_tcvt, dim3(D_DIM / 32, FSDIM / 32, 1), dim3(32, 8), 0, stream,
                     Sd, SdT, FSDIM, D_DIM);

  // shared expert gate+up -> Hs
  hipLaunchKernelGGL(moe_gateup, dim3(FSDIM / 64, T_TOK / 128, 1), dim3(256), 0, stream,
                     xb, SgT, SuT, Hs,
                     (const int*)nullptr, (const int*)nullptr, (const int*)nullptr,
                     FSDIM, D_DIM, D_DIM);
  // routed experts gate+up (gathered tokens) -> Hr (compacted rows)
  hipLaunchKernelGGL(moe_gateup, dim3(FDIM / 64, T_TOK / 128, NEXP), dim3(256), 0, stream,
                     xb, WgT, WuT, Hr, tokIdx, cnt, offs,
                     FDIM, D_DIM, D_DIM);
  // shared expert down: plain stores fully initialize out
  hipLaunchKernelGGL(moe_down, dim3(D_DIM / 64, T_TOK / 128, 1), dim3(256), 0, stream,
                     Hs, SdT, out,
                     (const int*)nullptr, (const int*)nullptr, (const int*)nullptr,
                     (const float*)nullptr, D_DIM, FSDIM);
  // routed experts down: weighted atomic scatter-add into out
  hipLaunchKernelGGL(moe_down, dim3(D_DIM / 64, T_TOK / 128, NEXP), dim3(256), 0, stream,
                     Hr, WdT, out, tokIdx, cnt, offs, tokW, D_DIM, FDIM);
}
